// MyGCNNet_89077621719480
// MI455X (gfx1250) — matrix operand order repaired and checked by reading, not compiled, not run on hardware
//
#include <hip/hip_runtime.h>

// ---------------------------------------------------------------------------
// MI455X / gfx1250 implementation.
//  * All dense GEMMs (GNN linears, convs 2-4 via implicit GEMM) run on
//    v_wmma_f32_16x16x32_f16 (f16 in, f32 accumulate).
//  * Activations are f16 everywhere to halve gather/scatter HBM traffic.
//  * Segment sums accumulate as f32 atomics into buffers sized to live in
//    the 192MB L2.
//  * BN/bias folded to per-channel scale/shift epilogues.
// Workspace budget ~0.8 GB (conv activations region is reused for GNN bufs).
// Nested param dicts are assumed flattened in jax-pytree (sorted-key) order.
// ---------------------------------------------------------------------------

typedef _Float16 h16;
typedef __attribute__((ext_vector_type(8)))  _Float16 v8h;
typedef __attribute__((ext_vector_type(16))) _Float16 v16h;
typedef __attribute__((ext_vector_type(8)))  float    v8f;

namespace {

constexpr int   kB    = 64;
constexpr int   kHW   = 128;      // conv1/2 spatial
constexpr int   kHW2  = 64;       // post-pool spatial
constexpr int   N_PX  = 262144;
constexpr int   E_PX  = 1048576;
constexpr int   N_SP  = 16384;
constexpr int   E_G   = 262144;
constexpr int   Dm    = 128;
constexpr int   NCls  = 10;
constexpr float kEPS  = 1e-5f;

// ========================= small prep / utility kernels ====================

__global__ void k_zero_f32(float* p, size_t n) {
  size_t i = (size_t)blockIdx.x * blockDim.x + threadIdx.x;
  if (i < n) p[i] = 0.f;
}

__global__ void k_f32_to_f16(const float* __restrict__ s, h16* __restrict__ d, int n) {
  int i = blockIdx.x * blockDim.x + threadIdx.x;
  if (i < n) d[i] = (h16)s[i];
}

// conv weight (O,I,3,3) f32 -> tap-major f16: d[(ty*3+tx)*O*I + o*I + i]
__global__ void k_convw_tap(const float* __restrict__ s, h16* __restrict__ d, int O, int I) {
  int e = blockIdx.x * blockDim.x + threadIdx.x;
  int n = O * I * 9;
  if (e >= n) return;
  int o   = e / (I * 9);
  int rem = e % (I * 9);
  int ci  = rem / 9;
  int ty  = (rem % 9) / 3;
  int tx  = rem % 3;
  d[(size_t)(ty * 3 + tx) * O * I + (size_t)o * I + ci] = (h16)s[e];
}

// fold conv bias + BN2d: y = conv*sc + sh
__global__ void k_bnfold_conv(const float* cb, const float* m, const float* v,
                              const float* g, const float* b,
                              float* sc, float* sh, int C) {
  int i = blockIdx.x * blockDim.x + threadIdx.x;
  if (i >= C) return;
  float k = g[i] * rsqrtf(v[i] + kEPS);
  sc[i] = k;
  sh[i] = (cb[i] - m[i]) * k + b[i];
}

// fold linear bias + BN1d: y = lin*sc + sh
__global__ void k_bnfold_lin(const float* b1, const float* m, const float* v,
                             const float* g, const float* b,
                             float* sc, float* sh, int C) {
  int i = blockIdx.x * blockDim.x + threadIdx.x;
  if (i >= C) return;
  float k = g[i] * rsqrtf(v[i] + kEPS);
  sc[i] = k;
  sh[i] = (b1[i] - m[i]) * k + b[i];
}

// ============================ WMMA GEMM engine =============================
// Out[M x 128] = A[M x 128] @ W^T  (W given row-major as W[n][k], f16)
// 256 threads = 8 waves; block computes a 128x128 output tile.
// Wave w owns rows [16w,16w+16) x all 128 cols (8 n-tiles).

enum EpiMode { EPI_BIAS = 0, EPI_GATHER_ADD = 1, EPI_SCATTER_ADD = 2,
               EPI_SCALE_SHIFT_RELU = 3, EPI_PLAIN = 4 };

template <int MODE>
__global__ __launch_bounds__(256) void gemm128(
    const h16* A, const h16* __restrict__ Wt, int M,
    h16* Out,                       // modes 0,1,3,4  (may alias A: in-place)
    float* __restrict__ AggOut,     // mode 2 scatter destination
    const float* __restrict__ p0,   // mode 0: bias ; mode 3: scale
    const float* __restrict__ p1,   // mode 3: shift
    const int* __restrict__ idx,    // mode 1: gather rows ; mode 2: scatter rows
    const h16* __restrict__ Xg)     // mode 1: gather source (N x 128)
{
  __shared__ h16 sW[128 * 128];
  __shared__ h16 sA[128 * 128];

  const int t  = threadIdx.x;
  const int m0 = blockIdx.x * 128;

  // ---- stage W (whole 128x128, f16) ----
  {
    const int row = t >> 1, cb = (t & 1) * 64;
    const v8h* src = (const v8h*)(Wt + (size_t)row * 128 + cb);
    v8h* dst = (v8h*)(sW + row * 128 + cb);
#pragma unroll
    for (int i = 0; i < 8; ++i) dst[i] = src[i];
  }
  // ---- stage A tile (zero-padded past M) ----
  {
    const int row = t >> 1, cb = (t & 1) * 64;
    const int gr = m0 + row;
    v8h* dst = (v8h*)(sA + row * 128 + cb);
    if (gr < M) {
      const v8h* src = (const v8h*)(A + (size_t)gr * 128 + cb);
#pragma unroll
      for (int i = 0; i < 8; ++i) dst[i] = src[i];
    } else {
      v8h z = {};
#pragma unroll
      for (int i = 0; i < 8; ++i) dst[i] = z;
    }
    if (m0 + 128 + row < M)  // warm L2 for the next tile
      __builtin_prefetch(A + (size_t)(m0 + 128 + row) * 128 + cb, 0, 0);
  }
  __syncthreads();

  const int wave = t >> 5, lane = t & 31;
  const int half = lane >> 4, l16 = lane & 15;
  const int mrow = wave * 16;

  v8f acc[8];
#pragma unroll
  for (int nt = 0; nt < 8; ++nt) acc[nt] = (v8f){};

#pragma unroll
  for (int kt = 0; kt < 4; ++kt) {
    // A fragment (16x32 f16): lanes0-15 K=kt*32+[0..7]/[16..23], lanes16-31 +8
    v16h af;
    {
      const h16* ap = sA + (mrow + l16) * 128 + kt * 32 + half * 8;
      v8h lo = *(const v8h*)ap;          // VGPRs 0..3
      v8h hi = *(const v8h*)(ap + 16);   // VGPRs 4..7
#pragma unroll
      for (int i = 0; i < 8; ++i) { af[i] = lo[i]; af[i + 8] = hi[i]; }
    }
#pragma unroll
    for (int nt = 0; nt < 8; ++nt) {
      // B fragment (32x16): lane n=l16, K=kt*32+half*8+[0..7]/[16..23] (matches A)
      v16h bf;
      {
        const h16* bp = sW + (nt * 16 + l16) * 128 + kt * 32 + half * 8;
        v8h lo = *(const v8h*)bp;
        v8h hi = *(const v8h*)(bp + 16);
#pragma unroll
        for (int i = 0; i < 8; ++i) { bf[i] = lo[i]; bf[i + 8] = hi[i]; }
      }
      acc[nt] = __builtin_amdgcn_wmma_f32_16x16x32_f16(
          false, af, false, bf, (short)0, acc[nt], false, false);
    }
  }

  // ---- epilogue: D VGPR r, lane -> (m = r + 8*half, n = l16) ----
#pragma unroll
  for (int nt = 0; nt < 8; ++nt) {
#pragma unroll
    for (int r = 0; r < 8; ++r) {
      const int gm = m0 + mrow + r + half * 8;
      const int gn = nt * 16 + l16;
      if (gm < M) {
        const float v = acc[nt][r];
        if constexpr (MODE == EPI_BIAS) {
          Out[(size_t)gm * 128 + gn] = (h16)(v + p0[gn]);
        } else if constexpr (MODE == EPI_GATHER_ADD) {
          const float xv = (float)Xg[(size_t)idx[gm] * 128 + gn];
          Out[(size_t)gm * 128 + gn] = (h16)(v + xv);
        } else if constexpr (MODE == EPI_SCATTER_ADD) {
          atomicAdd(AggOut + (size_t)idx[gm] * 128 + gn, v);
        } else if constexpr (MODE == EPI_SCALE_SHIFT_RELU) {
          const float y = v * p0[gn] + p1[gn];
          Out[(size_t)gm * 128 + gn] = (h16)(y > 0.f ? y : 0.f);
        } else {
          Out[(size_t)gm * 128 + gn] = (h16)v;
        }
      }
    }
  }
}

// ===================== WMMA implicit-GEMM 3x3 SAME conv ====================
// One block per (image b, output row h). M = HW pixels of the row,
// N = COUT, K = CIN per tap, accumulated over 9 taps.

template <int CIN, int COUT, int HW>
__global__ __launch_bounds__(256) void conv3x3_wmma(
    const h16* __restrict__ in, const h16* __restrict__ wtap,
    const float* __restrict__ sc, const float* __restrict__ sh,
    h16* __restrict__ out) {
  constexpr int NT = COUT / 16;
  constexpr int KT = CIN / 32;
  __shared__ h16 sA[HW * CIN];
  __shared__ h16 sW[COUT * CIN];

  const int t    = threadIdx.x;
  const int b    = blockIdx.x / HW;
  const int h    = blockIdx.x % HW;
  const int wave = t >> 5, lane = t & 31;
  const int half = lane >> 4, l16 = lane & 15;
  const int mt    = (wave * 4) / NT;   // constant per wave
  const int nbase = (wave * 4) % NT;

  v8f acc[4];
#pragma unroll
  for (int i = 0; i < 4; ++i) acc[i] = (v8f){};

  for (int tap = 0; tap < 9; ++tap) {
    const int dy = tap / 3 - 1, dx = tap % 3 - 1;
    __syncthreads();
    // stage im2col rows for this tap: sA[x][c]
    const int iy = h + dy;
    for (int e = t; e < HW * CIN; e += 256) {
      const int c = e / HW, x = e % HW;
      const int ix = x + dx;
      h16 v = (h16)0.f;
      if (iy >= 0 && iy < HW && ix >= 0 && ix < HW)
        v = in[(((size_t)b * CIN + c) * HW + iy) * HW + ix];
      sA[x * CIN + c] = v;
    }
    // stage this tap's weights: sW[o][k]
    const h16* wsrc = wtap + (size_t)tap * COUT * CIN;
    for (int e = t; e < COUT * CIN; e += 256) sW[e] = wsrc[e];
    __syncthreads();

#pragma unroll
    for (int kt = 0; kt < KT; ++kt) {
      v16h af;
      {
        const h16* ap = sA + (mt * 16 + l16) * CIN + kt * 32 + half * 8;
        v8h lo = *(const v8h*)ap;
        v8h hi = *(const v8h*)(ap + 16);
#pragma unroll
        for (int i = 0; i < 8; ++i) { af[i] = lo[i]; af[i + 8] = hi[i]; }
      }
#pragma unroll
      for (int ti = 0; ti < 4; ++ti) {
        const int n0 = (nbase + ti) * 16;
        v16h bf;
        {
          const h16* bp = sW + (n0 + l16) * CIN + kt * 32 + half * 8;
          v8h lo = *(const v8h*)bp;
          v8h hi = *(const v8h*)(bp + 16);
#pragma unroll
          for (int i = 0; i < 8; ++i) { bf[i] = lo[i]; bf[i + 8] = hi[i]; }
        }
        acc[ti] = __builtin_amdgcn_wmma_f32_16x16x32_f16(
            false, af, false, bf, (short)0, acc[ti], false, false);
      }
    }
  }

#pragma unroll
  for (int ti = 0; ti < 4; ++ti) {
    const int gn = (nbase + ti) * 16 + l16;
#pragma unroll
    for (int r = 0; r < 8; ++r) {
      const int xp = mt * 16 + r + half * 8;
      const float y = acc[ti][r] * sc[gn] + sh[gn];
      out[(((size_t)b * COUT + gn) * HW + h) * HW + xp] = (h16)(y > 0.f ? y : 0.f);
    }
  }
}

// =============================== conv1 (K=27) ==============================

__global__ __launch_bounds__(256) void k_conv1(
    const float* __restrict__ img, const float* __restrict__ w,
    const float* __restrict__ sc, const float* __restrict__ sh,
    h16* __restrict__ out) {
  __shared__ float sw[64 * 27];
  for (int e = threadIdx.x; e < 64 * 27; e += 256) sw[e] = w[e];
  __syncthreads();
  const size_t idx = (size_t)blockIdx.x * 256 + threadIdx.x;
  const int x = idx & 127;
  const int h = (idx >> 7) & 127;
  const int o = (idx >> 14) & 63;
  const int b = (int)(idx >> 20);
  float s = 0.f;
#pragma unroll
  for (int ci = 0; ci < 3; ++ci) {
#pragma unroll
    for (int ty = 0; ty < 3; ++ty) {
      const int iy = h + ty - 1;
      if (iy < 0 || iy >= 128) continue;
#pragma unroll
      for (int tx = 0; tx < 3; ++tx) {
        const int ix = x + tx - 1;
        if (ix < 0 || ix >= 128) continue;
        s += img[(((size_t)b * 3 + ci) * 128 + iy) * 128 + ix] *
             sw[(o * 3 + ci) * 9 + ty * 3 + tx];
      }
    }
  }
  const float y = s * sc[o] + sh[o];
  out[idx] = (h16)(y > 0.f ? y : 0.f);
}

// ============================== pool / gather ==============================

__global__ void k_maxpool2(const h16* __restrict__ in, h16* __restrict__ out) {
  const size_t idx = (size_t)blockIdx.x * 256 + threadIdx.x;   // B*64*64*64
  const int x = idx & 63;
  const int y = (idx >> 6) & 63;
  const int c = (idx >> 12) & 63;
  const int b = (int)(idx >> 18);
  const size_t base = (((size_t)b * 64 + c) * 128 + 2 * y) * 128 + 2 * x;
  float m = (float)in[base];
  m = fmaxf(m, (float)in[base + 1]);
  m = fmaxf(m, (float)in[base + 128]);
  m = fmaxf(m, (float)in[base + 129]);
  out[idx] = (h16)m;
}

__global__ void k_gather_px(const h16* __restrict__ act4,
                            const int* __restrict__ dw, h16* __restrict__ px) {
  const size_t idx = (size_t)blockIdx.x * 256 + threadIdx.x;   // N_PX*128
  const int i = (int)(idx >> 7), c = idx & 127;
  const int b = dw[i * 3 + 0], y = dw[i * 3 + 1], x = dw[i * 3 + 2];
  px[idx] = act4[(((size_t)b * 128 + c) * 64 + y) * 64 + x];
}

// ============================ GNN helper kernels ===========================

// pos1[e][d] = relu(ew[e][0]*pe_w1[d][0] + ew[e][1]*pe_w1[d][1])
__global__ void k_posemb1(const float* __restrict__ ew,
                          const float* __restrict__ pw1,
                          h16* __restrict__ out, int E) {
  const size_t idx = (size_t)blockIdx.x * 256 + threadIdx.x;
  if (idx >= (size_t)E * 128) return;
  const int e = (int)(idx >> 7), d = idx & 127;
  const float p = ew[(size_t)e * 2] * pw1[d * 2] + ew[(size_t)e * 2 + 1] * pw1[d * 2 + 1];
  out[idx] = (h16)(p > 0.f ? p : 0.f);
}

__global__ void k_count(const int* __restrict__ seg, float* cnt, int n) {
  const int i = blockIdx.x * 256 + threadIdx.x;
  if (i < n) atomicAdd(cnt + seg[i], 1.f);
}

__global__ void k_scatter_rows(const h16* __restrict__ h,
                               const int* __restrict__ seg,
                               float* sum, int rows) {
  const size_t idx = (size_t)blockIdx.x * 256 + threadIdx.x;
  if (idx >= (size_t)rows * 128) return;
  const int i = (int)(idx >> 7), c = idx & 127;
  atomicAdd(sum + (size_t)seg[i] * 128 + c, (float)h[idx]);
}

__global__ void k_divcnt_rows(const float* __restrict__ sum,
                              const float* __restrict__ cnt,
                              h16* __restrict__ out, int rows) {
  const size_t idx = (size_t)blockIdx.x * 256 + threadIdx.x;
  if (idx >= (size_t)rows * 128) return;
  out[idx] = (h16)(sum[idx] / fmaxf(cnt[idx >> 7], 1.f));
}

// ========================= attention / pool / readout ======================

__global__ void k_att(const h16* __restrict__ h2, const float* __restrict__ aw,
                      float* __restrict__ att, int n) {
  const int i = blockIdx.x * 256 + threadIdx.x;
  if (i >= n) return;
  float s = 0.f;
#pragma unroll 4
  for (int k = 0; k < 128; ++k) s += (float)h2[(size_t)i * 128 + k] * aw[k];
  att[i] = 1.f / (1.f + expf(-s));
}

__global__ void k_attpool(const h16* __restrict__ h2, const float* __restrict__ att,
                          const int* __restrict__ bg, float* __restrict__ hgsum,
                          int n) {
  const size_t idx = (size_t)blockIdx.x * 256 + threadIdx.x;
  if (idx >= (size_t)n * 128) return;
  const int i = (int)(idx >> 7), c = idx & 127;
  const float xa = (float)h2[idx] * (att[i] + 1.f) * 0.5f;
  atomicAdd(hgsum + (size_t)bg[i] * 128 + c, xa);
}

__global__ void k_div_hg(const float* __restrict__ hgsum,
                         const float* __restrict__ cnt, float* __restrict__ hg) {
  const int idx = blockIdx.x * 256 + threadIdx.x;   // 64*128
  if (idx < 64 * 128) hg[idx] = hgsum[idx] / fmaxf(cnt[idx >> 7], 1.f);
}

__global__ void k_readout(const float* __restrict__ hg,
                          const float* __restrict__ ro, float* __restrict__ out) {
  const int t = blockIdx.x * 256 + threadIdx.x;     // 64*10
  if (t >= 64 * NCls) return;
  const int b = t / NCls, c = t % NCls;
  float s = 0.f;
#pragma unroll 4
  for (int k = 0; k < 128; ++k) s += hg[b * 128 + k] * ro[c * 128 + k];
  out[t] = s;
}

// ================================ host side ================================

struct BNp { const float *b, *g, *m, *v; };
struct LPp { const float* b1; BNp bn; const float *pe_w1, *pe_w2, *w1, *w2; };
struct GLayer { h16 *pew2, *w1h, *w2h; const float* pe_w1; float *sc, *sh; };

inline unsigned cdiv(size_t n, size_t d) { return (unsigned)((n + d - 1) / d); }

static void run_gnn(hipStream_t s, int Nn, int Ne,
                    const int* ei, const float* ew,
                    const h16* embw, const float* embb,
                    const GLayer* L, int nl,
                    const h16* x0, h16* hA, h16* hB, h16* ebuf,
                    float* agg, h16* aggm, const float* cnt, h16** result) {
  const dim3 blk(256);
  // embedding: hA = x0 @ emb_w^T + emb_b
  gemm128<EPI_BIAS><<<cdiv(Nn, 128), blk, 0, s>>>(
      x0, embw, Nn, hA, nullptr, embb, nullptr, nullptr, nullptr);
  h16* cur = hA; h16* nxt = hB;
  for (int l = 0; l < nl; ++l) {
    // pos1 = relu(ew @ pe_w1^T)           (into ebuf)
    k_posemb1<<<cdiv((size_t)Ne * 128, 256), blk, 0, s>>>(ew, L[l].pe_w1, ebuf, Ne);
    // einp = pos1 @ pe_w2^T + cur[src]    (in place)
    gemm128<EPI_GATHER_ADD><<<cdiv(Ne, 128), blk, 0, s>>>(
        ebuf, L[l].pew2, Ne, ebuf, nullptr, nullptr, nullptr, ei, cur);
    // agg = segment_sum(einp @ w2^T, dst)
    k_zero_f32<<<cdiv((size_t)Nn * 128, 256), blk, 0, s>>>(agg, (size_t)Nn * 128);
    gemm128<EPI_SCATTER_ADD><<<cdiv(Ne, 128), blk, 0, s>>>(
        ebuf, L[l].w2h, Ne, nullptr, agg, nullptr, nullptr, ei + Ne, nullptr);
    // mean
    k_divcnt_rows<<<cdiv((size_t)Nn * 128, 256), blk, 0, s>>>(agg, cnt, aggm, Nn);
    // h = relu(bn(agg_mean @ w1^T + b1))  (folded scale/shift)
    gemm128<EPI_SCALE_SHIFT_RELU><<<cdiv(Nn, 128), blk, 0, s>>>(
        aggm, L[l].w1h, Nn, nxt, nullptr, L[l].sc, L[l].sh, nullptr, nullptr);
    h16* tmp = cur; cur = nxt; nxt = tmp;
  }
  *result = cur;
}

} // namespace

extern "C" void kernel_launch(void* const* d_in, const int* in_sizes, int n_in,
                              void* d_out, int out_size, void* d_ws, size_t ws_size,
                              hipStream_t stream) {
  (void)in_sizes; (void)n_in; (void)out_size; (void)ws_size;
  const dim3 blk(256);

  // ---------------- unpack inputs (pytree sorted-key order) ----------------
  int ip = 0;
  auto F = [&]() { return (const float*)d_in[ip++]; };
  auto I = [&]() { return (const int*)d_in[ip++]; };
  const float* images    = F();
  const int*   data_where= I();
  const int*   ei_px     = I();
  const float* ew_px     = F();
  const int*   batch_px  = I();
  const int*   ei_g      = I();
  const float* ew_g      = F();
  const int*   batch_g   = I();
  auto getbn = [&]() { BNp r; r.b = F(); r.g = F(); r.m = F(); r.v = F(); return r; };
  BNp bn1 = getbn(), bn2 = getbn(), bn3 = getbn(), bn4 = getbn();
  const float* c1_b = F(); const float* c1_w = F();
  const float* c2_b = F(); const float* c2_w = F();
  const float* c3_b = F(); const float* c3_w = F();
  const float* c4_b = F(); const float* c4_w = F();
  auto getlp = [&]() { LPp r; r.b1 = F(); r.bn = getbn();
                       r.pe_w1 = F(); r.pe_w2 = F(); r.w1 = F(); r.w2 = F(); return r; };
  const float* g1_emb_b = F(); const float* g1_emb_w = F();
  LPp g1L[2] = { getlp(), getlp() };
  const float* g2_emb_b = F(); const float* g2_emb_w = F();
  LPp g2L[4] = { getlp(), getlp(), getlp(), getlp() };
  const float* att_w = F(); const float* ro_w = F();
  float* outp = (float*)d_out;

  // ---------------------------- workspace carve ----------------------------
  char* wsp = (char*)d_ws;
  auto alloc = [&](size_t bytes) { char* r = wsp; wsp += (bytes + 255) & ~(size_t)255; return r; };

  // persistent: f16 weights + folded params
  h16* c2w_h = (h16*)alloc((size_t)64 * 64 * 9 * 2);
  h16* c3w_h = (h16*)alloc((size_t)128 * 64 * 9 * 2);
  h16* c4w_h = (h16*)alloc((size_t)128 * 128 * 9 * 2);
  float *cs1=(float*)alloc(64*4),  *ct1=(float*)alloc(64*4);
  float *cs2=(float*)alloc(64*4),  *ct2=(float*)alloc(64*4);
  float *cs3=(float*)alloc(128*4), *ct3=(float*)alloc(128*4);
  float *cs4=(float*)alloc(128*4), *ct4=(float*)alloc(128*4);
  h16* g1embw_h = (h16*)alloc((size_t)Dm * Dm * 2);
  h16* g2embw_h = (h16*)alloc((size_t)Dm * Dm * 2);
  GLayer G1[2], G2[4];
  for (int l = 0; l < 2; ++l) {
    G1[l].pew2 = (h16*)alloc((size_t)Dm*Dm*2); G1[l].w1h = (h16*)alloc((size_t)Dm*Dm*2);
    G1[l].w2h  = (h16*)alloc((size_t)Dm*Dm*2);
    G1[l].sc = (float*)alloc(Dm*4); G1[l].sh = (float*)alloc(Dm*4);
    G1[l].pe_w1 = g1L[l].pe_w1;
  }
  for (int l = 0; l < 4; ++l) {
    G2[l].pew2 = (h16*)alloc((size_t)Dm*Dm*2); G2[l].w1h = (h16*)alloc((size_t)Dm*Dm*2);
    G2[l].w2h  = (h16*)alloc((size_t)Dm*Dm*2);
    G2[l].sc = (float*)alloc(Dm*4); G2[l].sh = (float*)alloc(Dm*4);
    G2[l].pe_w1 = g2L[l].pe_w1;
  }
  // survives both phases
  h16* px = (h16*)alloc((size_t)N_PX * 128 * 2);
  char* mark = wsp;

  // phase-1 region: conv activations
  h16* act1 = (h16*)alloc((size_t)kB * 64 * kHW * kHW * 2);
  h16* act2 = (h16*)alloc((size_t)kB * 64 * kHW * kHW * 2);
  h16* actp = (h16*)alloc((size_t)kB * 64 * kHW2 * kHW2 * 2);
  h16* act3 = (h16*)alloc((size_t)kB * 128 * kHW2 * kHW2 * 2);
  h16* act4 = (h16*)alloc((size_t)kB * 128 * kHW2 * kHW2 * 2);

  // ------------------------------ prep phase -------------------------------
  auto conv_f16 = [&](const float* s, h16* d, int O, int Ii) {
    k_convw_tap<<<cdiv((size_t)O * Ii * 9, 256), blk, 0, stream>>>(s, d, O, Ii);
  };
  conv_f16(c2_w, c2w_h, 64, 64);
  conv_f16(c3_w, c3w_h, 128, 64);
  conv_f16(c4_w, c4w_h, 128, 128);
  k_bnfold_conv<<<1, 128, 0, stream>>>(c1_b, bn1.m, bn1.v, bn1.g, bn1.b, cs1, ct1, 64);
  k_bnfold_conv<<<1, 128, 0, stream>>>(c2_b, bn2.m, bn2.v, bn2.g, bn2.b, cs2, ct2, 64);
  k_bnfold_conv<<<1, 128, 0, stream>>>(c3_b, bn3.m, bn3.v, bn3.g, bn3.b, cs3, ct3, 128);
  k_bnfold_conv<<<1, 128, 0, stream>>>(c4_b, bn4.m, bn4.v, bn4.g, bn4.b, cs4, ct4, 128);
  k_f32_to_f16<<<cdiv(Dm*Dm, 256), blk, 0, stream>>>(g1_emb_w, g1embw_h, Dm*Dm);
  k_f32_to_f16<<<cdiv(Dm*Dm, 256), blk, 0, stream>>>(g2_emb_w, g2embw_h, Dm*Dm);
  for (int l = 0; l < 2; ++l) {
    k_f32_to_f16<<<cdiv(Dm*Dm,256), blk, 0, stream>>>(g1L[l].pe_w2, G1[l].pew2, Dm*Dm);
    k_f32_to_f16<<<cdiv(Dm*Dm,256), blk, 0, stream>>>(g1L[l].w1,    G1[l].w1h,  Dm*Dm);
    k_f32_to_f16<<<cdiv(Dm*Dm,256), blk, 0, stream>>>(g1L[l].w2,    G1[l].w2h,  Dm*Dm);
    k_bnfold_lin<<<1, 128, 0, stream>>>(g1L[l].b1, g1L[l].bn.m, g1L[l].bn.v,
                                        g1L[l].bn.g, g1L[l].bn.b, G1[l].sc, G1[l].sh, Dm);
  }
  for (int l = 0; l < 4; ++l) {
    k_f32_to_f16<<<cdiv(Dm*Dm,256), blk, 0, stream>>>(g2L[l].pe_w2, G2[l].pew2, Dm*Dm);
    k_f32_to_f16<<<cdiv(Dm*Dm,256), blk, 0, stream>>>(g2L[l].w1,    G2[l].w1h,  Dm*Dm);
    k_f32_to_f16<<<cdiv(Dm*Dm,256), blk, 0, stream>>>(g2L[l].w2,    G2[l].w2h,  Dm*Dm);
    k_bnfold_lin<<<1, 128, 0, stream>>>(g2L[l].b1, g2L[l].bn.m, g2L[l].bn.v,
                                        g2L[l].bn.g, g2L[l].bn.b, G2[l].sc, G2[l].sh, Dm);
  }

  // ------------------------------- conv net --------------------------------
  k_conv1<<<cdiv((size_t)kB * 64 * kHW * kHW, 256), blk, 0, stream>>>(
      images, c1_w, cs1, ct1, act1);
  conv3x3_wmma<64, 64, 128><<<kB * kHW, blk, 0, stream>>>(act1, c2w_h, cs2, ct2, act2);
  k_maxpool2<<<cdiv((size_t)kB * 64 * kHW2 * kHW2, 256), blk, 0, stream>>>(act2, actp);
  conv3x3_wmma<64, 128, 64><<<kB * kHW2, blk, 0, stream>>>(actp, c3w_h, cs3, ct3, act3);
  conv3x3_wmma<128, 128, 64><<<kB * kHW2, blk, 0, stream>>>(act3, c4w_h, cs4, ct4, act4);
  k_gather_px<<<cdiv((size_t)N_PX * 128, 256), blk, 0, stream>>>(act4, data_where, px);

  // -------- phase-2 region: reuse conv-activation memory for GNN bufs ------
  wsp = mark;
  h16*  ebuf  = (h16*)alloc((size_t)E_PX * 128 * 2);
  float* agg  = (float*)alloc((size_t)N_PX * 128 * 4);
  h16*  aggm  = (h16*)alloc((size_t)N_PX * 128 * 2);
  h16*  hA    = (h16*)alloc((size_t)N_PX * 128 * 2);
  h16*  hB    = (h16*)alloc((size_t)N_PX * 128 * 2);
  float* cntpx= (float*)alloc((size_t)N_PX * 4);
  float* spsum= (float*)alloc((size_t)N_SP * 128 * 4);
  float* cntsp= (float*)alloc((size_t)N_SP * 4);
  h16*  sp    = (h16*)alloc((size_t)N_SP * 128 * 2);
  h16*  ebufg = (h16*)alloc((size_t)E_G * 128 * 2);
  float* aggg = (float*)alloc((size_t)N_SP * 128 * 4);
  h16*  aggmg = (h16*)alloc((size_t)N_SP * 128 * 2);
  h16*  hAg   = (h16*)alloc((size_t)N_SP * 128 * 2);
  h16*  hBg   = (h16*)alloc((size_t)N_SP * 128 * 2);
  float* cntg = (float*)alloc((size_t)N_SP * 4);
  float* attb = (float*)alloc((size_t)N_SP * 4);
  float* hgsum= (float*)alloc((size_t)64 * 128 * 4);
  float* cntb = (float*)alloc((size_t)64 * 4);
  float* hg   = (float*)alloc((size_t)64 * 128 * 4);

  // ------------------------------ pixel GNN --------------------------------
  k_zero_f32<<<cdiv(N_PX, 256), blk, 0, stream>>>(cntpx, N_PX);
  k_count<<<cdiv(E_PX, 256), blk, 0, stream>>>(ei_px + E_PX, cntpx, E_PX);
  h16* h1 = nullptr;
  run_gnn(stream, N_PX, E_PX, ei_px, ew_px, g1embw_h, g1_emb_b, G1, 2,
          px, hA, hB, ebuf, agg, aggm, cntpx, &h1);

  // superpixel pooling
  k_zero_f32<<<cdiv((size_t)N_SP * 128, 256), blk, 0, stream>>>(spsum, (size_t)N_SP * 128);
  k_zero_f32<<<cdiv(N_SP, 256), blk, 0, stream>>>(cntsp, N_SP);
  k_count<<<cdiv(N_PX, 256), blk, 0, stream>>>(batch_px, cntsp, N_PX);
  k_scatter_rows<<<cdiv((size_t)N_PX * 128, 256), blk, 0, stream>>>(h1, batch_px, spsum, N_PX);
  k_divcnt_rows<<<cdiv((size_t)N_SP * 128, 256), blk, 0, stream>>>(spsum, cntsp, sp, N_SP);

  // ---------------------------- superpixel GNN -----------------------------
  k_zero_f32<<<cdiv(N_SP, 256), blk, 0, stream>>>(cntg, N_SP);
  k_count<<<cdiv(E_G, 256), blk, 0, stream>>>(ei_g + E_G, cntg, E_G);
  h16* h2 = nullptr;
  run_gnn(stream, N_SP, E_G, ei_g, ew_g, g2embw_h, g2_emb_b, G2, 4,
          sp, hAg, hBg, ebufg, aggg, aggmg, cntg, &h2);

  // --------------------- attention pooling + readout -----------------------
  k_att<<<cdiv(N_SP, 256), blk, 0, stream>>>(h2, att_w, attb, N_SP);
  k_zero_f32<<<cdiv(64 * 128, 256), blk, 0, stream>>>(hgsum, 64 * 128);
  k_zero_f32<<<1, 64, 0, stream>>>(cntb, 64);
  k_count<<<cdiv(N_SP, 256), blk, 0, stream>>>(batch_g, cntb, N_SP);
  k_attpool<<<cdiv((size_t)N_SP * 128, 256), blk, 0, stream>>>(h2, attb, batch_g, hgsum, N_SP);
  k_div_hg<<<cdiv(64 * 128, 256), blk, 0, stream>>>(hgsum, cntb, hg);
  k_readout<<<cdiv(64 * NCls, 256), blk, 0, stream>>>(hg, ro_w, outp);
}
